// GAT_2688649527898
// MI455X (gfx1250) — matrix operand order repaired and checked by reading, not compiled, not run on hardware
//
#include <hip/hip_runtime.h>
#include <hip/hip_bf16.h>
#include <stdint.h>

// ---------------------------------------------------------------------------
// GAT forward for MI455X (gfx1250): bf16 WMMA, flash-style streamed softmax,
// adjacency read once per layer and cached as LDS bitmask.
// Fragment loads use contiguous 32B/16B vector loads (k = 16*half + s order, same in A and B).
// ---------------------------------------------------------------------------

typedef __bf16 bf16_t;
typedef __attribute__((ext_vector_type(16))) __bf16 v16bf;
typedef __attribute__((ext_vector_type(8)))  __bf16 v8bf;
typedef __attribute__((ext_vector_type(8)))  float  v8f;

#define GN   4096
#define GF   256
#define GHID 64
#define GH   8
#define GC   41
#define GCP  48      // padded classes (3 x 16)
#define GF2  512     // H*HID
#define ALPHA 0.2f

__device__ __forceinline__ bf16_t f2b(float f) { return (bf16_t)f; }  // native v_cvt

__device__ __forceinline__ v8f wmma_bf16(v16bf a, v16bf b, v8f c) {
  return __builtin_amdgcn_wmma_f32_16x16x32_bf16(false, a, false, b,
                                                 (short)0, c, false, false);
}

// A-fragment K offset for 16-bit 16x32 A (contiguous: k = 16*half + 2g): VGPR g, lane-half.
__device__ __forceinline__ int kA(int g, int half) {
  return 2 * g + half * 16;
}

__device__ __forceinline__ float lrelu(float x) { return x > 0.f ? x : ALPHA * x; }
__device__ __forceinline__ float eluf(float x)  { return x > 0.f ? x : __expf(x) - 1.f; }

// ===========================================================================
// K1: Wh[h] = x @ W[h]  (bf16 WMMA, fp32 accum). Emits Whb (bf16, [h][64][N]
// column-major for B-fragment loads) and f1/f2 = Wh @ a halves.
// grid (N/16, H), block 128 (4 waves; wave w owns N-tile w).
// ===========================================================================
__global__ __launch_bounds__(128) void gat_gemm1(
    const float* __restrict__ x, const float* __restrict__ W,
    const float* __restrict__ a, bf16_t* __restrict__ Whb,
    float* __restrict__ f1, float* __restrict__ f2) {
  const int rowBase = blockIdx.x * 16;
  const int h    = blockIdx.y;
  const int tid  = threadIdx.x;
  const int wave = tid >> 5, lane = tid & 31;
  const int m    = lane & 15, half = lane >> 4;

  __shared__ __align__(32) bf16_t Bcm[64][32];   // W^T tile: Bcm[n][k]
  __shared__ float f1acc[16], f2acc[16];
  if (tid < 16) { f1acc[tid] = 0.f; f2acc[tid] = 0.f; }

  v8f acc = {};
  const float* Wh_ = W + (size_t)h * GF * GHID;

  for (int kb = 0; kb < GF; kb += 32) {
    __syncthreads();
    for (int idx = tid; idx < 64 * 32; idx += 128) {
      int n = idx >> 5, k = idx & 31;
      Bcm[n][k] = f2b(Wh_[(size_t)(kb + k) * GHID + n]);
    }
    __syncthreads();
    // A fragment: two contiguous float4-pairs per lane, fp32 -> bf16.
    const float* xr = x + (size_t)(rowBase + m) * GF + kb;
    const float4* xc0 = (const float4*)(xr + half * 16);
    const float4* xc1 = (const float4*)(xr + half * 16 + 8);
    float4 c0 = xc0[0], c1 = xc0[1], d0 = xc1[0], d1 = xc1[1];
    v16bf afrag;
    afrag[0]  = f2b(c0.x); afrag[1]  = f2b(c0.y); afrag[2]  = f2b(c0.z); afrag[3]  = f2b(c0.w);
    afrag[4]  = f2b(c1.x); afrag[5]  = f2b(c1.y); afrag[6]  = f2b(c1.z); afrag[7]  = f2b(c1.w);
    afrag[8]  = f2b(d0.x); afrag[9]  = f2b(d0.y); afrag[10] = f2b(d0.z); afrag[11] = f2b(d0.w);
    afrag[12] = f2b(d1.x); afrag[13] = f2b(d1.y); afrag[14] = f2b(d1.z); afrag[15] = f2b(d1.w);
    // B fragment: one contiguous 32B LDS read (K = half*16 .. +15).
    v16bf bfrag = *(const v16bf*)&Bcm[wave * 16 + m][half * 16];
    acc = wmma_bf16(afrag, bfrag, acc);
  }

  // Epilogue: lane holds (row = g + 8*half, col n = wave*16 + m)
  const float* a1 = a + (size_t)h * 2 * GHID;
  const float* a2 = a1 + GHID;
  const int n = wave * 16 + m;
#pragma unroll
  for (int g = 0; g < 8; ++g) {
    float v = acc[g];
    int row = g + 8 * half;
    Whb[((size_t)h * GHID + n) * GN + rowBase + row] = f2b(v);
    atomicAdd(&f1acc[row], v * a1[n]);
    atomicAdd(&f2acc[row], v * a2[n]);
  }
  __syncthreads();
  if (tid < 16) {
    f1[(size_t)h * GN + rowBase + tid] = f1acc[tid];
    f2[(size_t)h * GN + rowBase + tid] = f2acc[tid];
  }
}

// ===========================================================================
// K2: fused masked-softmax attention + aggregation, layer 1, all 8 heads per
// block (adjacency read ONCE per 16-row block, cached as LDS bitmask).
// grid N/16, block 256 (wave = head). Output: hcat bf16 [N][512] (post-ELU).
// ===========================================================================
__global__ __launch_bounds__(256) void gat_attn1(
    const int* __restrict__ adj, const bf16_t* __restrict__ Whb,
    const float* __restrict__ f1, const float* __restrict__ f2,
    bf16_t* __restrict__ hcat) {
  const int rowBase = blockIdx.x * 16;
  const int tid  = threadIdx.x;
  const int h    = tid >> 5;          // wave == head
  const int lane = tid & 31;
  const int m    = lane & 15, half = lane >> 4;

  __shared__ uint32_t maskb[16][128];  // 4096-bit adjacency per row

  // Build bitmask: 2048 dwords, 8 per thread, 128-byte int4 loads.
  for (int it = 0; it < 8; ++it) {
    int idx = tid + it * 256;
    int r = idx >> 7, dw = idx & 127;
    const int4* ap = (const int4*)(adj + (size_t)(rowBase + r) * GN + dw * 32);
    uint32_t bits = 0;
#pragma unroll
    for (int q = 0; q < 8; ++q) {
      int4 v = ap[q];
      bits |= (v.x > 0 ? 1u : 0u) << (q * 4 + 0);
      bits |= (v.y > 0 ? 1u : 0u) << (q * 4 + 1);
      bits |= (v.z > 0 ? 1u : 0u) << (q * 4 + 2);
      bits |= (v.w > 0 ? 1u : 0u) << (q * 4 + 3);
    }
    maskb[r][dw] = bits;
  }
  __syncthreads();

  const float* f2h = f2 + (size_t)h * GN;
  // Masked row max of f2 (lrelu is monotone => gives softmax row max cheaply).
  float mx = -3.0e38f;
  {
    int start = half * 64;
    for (int dw = start; dw < start + 64; ++dw) {
      uint32_t bits = maskb[m][dw];
      if (!bits) continue;
      const float* fj = f2h + dw * 32;
#pragma unroll 8
      for (int b = 0; b < 32; ++b) {
        float v = ((bits >> b) & 1u) ? fj[b] : -3.0e38f;
        mx = fmaxf(mx, v);
      }
    }
  }
  mx = fmaxf(mx, __shfl_xor(mx, 16));
  const float f1m  = f1[(size_t)h * GN + rowBase + m];
  const float emax = lrelu(f1m + mx);

  v8f acc[4]; acc[0] = v8f{}; acc[1] = v8f{}; acc[2] = v8f{}; acc[3] = v8f{};
  float psum = 0.f;
  const bf16_t* WhbH = Whb + (size_t)h * GHID * GN;

  for (int jt = 0; jt < 128; ++jt) {        // 128 K-tiles of 32 columns
    const uint32_t bits = maskb[m][jt];
    const float* fj = f2h + jt * 32;
    v16bf p;
#pragma unroll
    for (int g = 0; g < 8; ++g) {
      int k = kA(g, half);
      float e0 = lrelu(f1m + fj[k])     - emax;
      float e1 = lrelu(f1m + fj[k + 1]) - emax;
      float p0 = ((bits >> k) & 1u)       ? __expf(e0) : 0.f;
      float p1 = ((bits >> (k + 1)) & 1u) ? __expf(e1) : 0.f;
      psum += p0 + p1;
      p[2 * g]     = f2b(p0);
      p[2 * g + 1] = f2b(p1);
    }
#pragma unroll
    for (int nt = 0; nt < 4; ++nt) {
      // Whole B fragment is one contiguous 32B region of column-major Whb.
      v16bf b = *(const v16bf*)(WhbH + (size_t)(nt * 16 + m) * GN + jt * 32 + half * 16);
      acc[nt] = wmma_bf16(p, b, acc[nt]);
    }
  }

  // Normalize by streamed row sum, ELU, store concat features.
  psum += __shfl_xor(psum, 16);             // lane r holds row r's full sum
#pragma unroll
  for (int g = 0; g < 8; ++g) {
    int row = g + 8 * half;
    float inv = 1.f / __shfl(psum, row);
#pragma unroll
    for (int nt = 0; nt < 4; ++nt) {
      float v = eluf(acc[nt][g] * inv);
      hcat[(size_t)(rowBase + row) * GF2 + h * GHID + nt * 16 + m] = f2b(v);
    }
  }
}

// ===========================================================================
// K3: Wh2 = xc @ W_out (512 x 41, padded to 48). grid N/16, block 96 (3 waves).
// ===========================================================================
__global__ __launch_bounds__(96) void gat_gemm2(
    const bf16_t* __restrict__ hcat, const float* __restrict__ Wout,
    const float* __restrict__ aout, bf16_t* __restrict__ Wh2b,
    float* __restrict__ f1o, float* __restrict__ f2o) {
  const int rowBase = blockIdx.x * 16;
  const int tid  = threadIdx.x;
  const int wave = tid >> 5, lane = tid & 31;
  const int m    = lane & 15, half = lane >> 4;

  __shared__ __align__(32) bf16_t Bcm[GCP][32];
  __shared__ float f1acc[16], f2acc[16];
  if (tid < 16) { f1acc[tid] = 0.f; f2acc[tid] = 0.f; }

  v8f acc = {};
  for (int kb = 0; kb < GF2; kb += 32) {
    __syncthreads();
    for (int idx = tid; idx < GCP * 32; idx += 96) {
      int n = idx >> 5, k = idx & 31;
      Bcm[n][k] = (n < GC) ? f2b(Wout[(size_t)(kb + k) * GC + n]) : f2b(0.f);
    }
    __syncthreads();
    // A fragment: two contiguous 16B bf16 chunks per lane.
    const bf16_t* xr = hcat + (size_t)(rowBase + m) * GF2 + kb;
    v8bf lo = *(const v8bf*)(xr + half * 16);
    v8bf hi = *(const v8bf*)(xr + half * 16 + 8);
    v16bf afrag;
#pragma unroll
    for (int g = 0; g < 8; ++g) { afrag[g] = lo[g]; afrag[8 + g] = hi[g]; }
    v16bf bfrag = *(const v16bf*)&Bcm[wave * 16 + m][half * 16];
    acc = wmma_bf16(afrag, bfrag, acc);
  }

  const int n = wave * 16 + m;
  const float an1 = (n < GC) ? aout[n]      : 0.f;
  const float an2 = (n < GC) ? aout[GC + n] : 0.f;
#pragma unroll
  for (int g = 0; g < 8; ++g) {
    float v = acc[g];
    int row = g + 8 * half;
    Wh2b[(size_t)n * GN + rowBase + row] = f2b(v);   // cols >=41 are zero
    atomicAdd(&f1acc[row], v * an1);
    atomicAdd(&f2acc[row], v * an2);
  }
  __syncthreads();
  if (tid < 16) { f1o[rowBase + tid] = f1acc[tid]; f2o[rowBase + tid] = f2acc[tid]; }
}

// ===========================================================================
// K4: layer-2 attention + ELU + row log-softmax. grid N/16, block 128
// (4 waves split the 4096-j range; partials merged in LDS).
// ===========================================================================
__global__ __launch_bounds__(128) void gat_attn2(
    const int* __restrict__ adj, const bf16_t* __restrict__ Wh2b,
    const float* __restrict__ f1o, const float* __restrict__ f2o,
    float* __restrict__ out) {
  const int rowBase = blockIdx.x * 16;
  const int tid  = threadIdx.x;
  const int wave = tid >> 5, lane = tid & 31;
  const int m    = lane & 15, half = lane >> 4;

  __shared__ uint32_t maskb[16][128];
  __shared__ float hbuf[16][GCP];
  __shared__ float rsum[16];
  __shared__ float wmx[4][16];
  __shared__ float gmx[16];

  for (int it = 0; it < 16; ++it) {
    int idx = tid + it * 128;
    int r = idx >> 7, dw = idx & 127;
    const int4* ap = (const int4*)(adj + (size_t)(rowBase + r) * GN + dw * 32);
    uint32_t bits = 0;
#pragma unroll
    for (int q = 0; q < 8; ++q) {
      int4 v = ap[q];
      bits |= (v.x > 0 ? 1u : 0u) << (q * 4 + 0);
      bits |= (v.y > 0 ? 1u : 0u) << (q * 4 + 1);
      bits |= (v.z > 0 ? 1u : 0u) << (q * 4 + 2);
      bits |= (v.w > 0 ? 1u : 0u) << (q * 4 + 3);
    }
    maskb[r][dw] = bits;
  }
  for (int i = tid; i < 16 * GCP; i += 128) hbuf[i / GCP][i % GCP] = 0.f;
  if (tid < 16) rsum[tid] = 0.f;
  __syncthreads();

  // masked max of f2o over this wave's quarter of j
  float mx = -3.0e38f;
  {
    int start = wave * 32 + half * 16;
    for (int dw = start; dw < start + 16; ++dw) {
      uint32_t bits = maskb[m][dw];
      if (!bits) continue;
      const float* fj = f2o + dw * 32;
#pragma unroll 8
      for (int b = 0; b < 32; ++b) {
        float v = ((bits >> b) & 1u) ? fj[b] : -3.0e38f;
        mx = fmaxf(mx, v);
      }
    }
  }
  mx = fmaxf(mx, __shfl_xor(mx, 16));
  if (lane < 16) wmx[wave][lane] = mx;
  __syncthreads();
  if (tid < 16)
    gmx[tid] = fmaxf(fmaxf(wmx[0][tid], wmx[1][tid]),
                     fmaxf(wmx[2][tid], wmx[3][tid]));
  __syncthreads();

  const float f1m  = f1o[rowBase + m];
  const float emax = lrelu(f1m + gmx[m]);

  v8f acc[3]; acc[0] = v8f{}; acc[1] = v8f{}; acc[2] = v8f{};
  float psum = 0.f;
  for (int jt = wave * 32; jt < wave * 32 + 32; ++jt) {
    const uint32_t bits = maskb[m][jt];
    const float* fj = f2o + jt * 32;
    v16bf p;
#pragma unroll
    for (int g = 0; g < 8; ++g) {
      int k = kA(g, half);
      float e0 = lrelu(f1m + fj[k])     - emax;
      float e1 = lrelu(f1m + fj[k + 1]) - emax;
      float p0 = ((bits >> k) & 1u)       ? __expf(e0) : 0.f;
      float p1 = ((bits >> (k + 1)) & 1u) ? __expf(e1) : 0.f;
      psum += p0 + p1;
      p[2 * g] = f2b(p0); p[2 * g + 1] = f2b(p1);
    }
#pragma unroll
    for (int nt = 0; nt < 3; ++nt) {
      v16bf b = *(const v16bf*)(Wh2b + (size_t)(nt * 16 + m) * GN + jt * 32 + half * 16);
      acc[nt] = wmma_bf16(p, b, acc[nt]);
    }
  }

  psum += __shfl_xor(psum, 16);
  if (lane < 16) atomicAdd(&rsum[lane], psum);
#pragma unroll
  for (int g = 0; g < 8; ++g) {
    int row = g + 8 * half;
#pragma unroll
    for (int nt = 0; nt < 3; ++nt)
      atomicAdd(&hbuf[row][nt * 16 + m], acc[nt][g]);
  }
  __syncthreads();

  // ELU + log-softmax over the 41 classes, one thread per row.
  if (tid < 16) {
    int row = tid;
    float inv = 1.f / rsum[row];
    float vmax = -3.0e38f;
    for (int c = 0; c < GC; ++c) {
      float v = eluf(hbuf[row][c] * inv);
      hbuf[row][c] = v;
      vmax = fmaxf(vmax, v);
    }
    float s = 0.f;
    for (int c = 0; c < GC; ++c) s += __expf(hbuf[row][c] - vmax);
    float lse = vmax + __logf(s);
    for (int c = 0; c < GC; ++c)
      out[(size_t)(rowBase + row) * GC + c] = hbuf[row][c] - lse;
  }
}

// ===========================================================================
extern "C" void kernel_launch(void* const* d_in, const int* in_sizes, int n_in,
                              void* d_out, int out_size, void* d_ws, size_t ws_size,
                              hipStream_t stream) {
  const float* x    = (const float*)d_in[0];
  const int*   adj  = (const int*)  d_in[1];
  const float* W    = (const float*)d_in[2];
  const float* a    = (const float*)d_in[3];
  const float* Wout = (const float*)d_in[4];
  const float* aout = (const float*)d_in[5];
  float* out = (float*)d_out;

  char* ws = (char*)d_ws;
  size_t off = 0;
  bf16_t* Whb  = (bf16_t*)(ws + off); off += (size_t)GH * GHID * GN * 2;  // 4 MiB
  float*  f1   = (float*) (ws + off); off += (size_t)GH * GN * 4;
  float*  f2   = (float*) (ws + off); off += (size_t)GH * GN * 4;
  bf16_t* hcat = (bf16_t*)(ws + off); off += (size_t)GN * GF2 * 2;        // 4 MiB
  bf16_t* Wh2b = (bf16_t*)(ws + off); off += (size_t)GCP * GN * 2;
  float*  f1o  = (float*) (ws + off); off += (size_t)GN * 4;
  float*  f2o  = (float*) (ws + off); off += (size_t)GN * 4;

  dim3 g1(GN / 16, GH);
  gat_gemm1<<<g1, 128, 0, stream>>>(x, W, a, Whb, f1, f2);
  gat_attn1<<<GN / 16, 256, 0, stream>>>(adj, Whb, f1, f2, hcat);
  gat_gemm2<<<GN / 16, 96, 0, stream>>>(hcat, Wout, aout, Wh2b, f1o, f2o);
  gat_attn2<<<GN / 16, 128, 0, stream>>>(adj, Wh2b, f1o, f2o, out);
}
